// MAGNOLayer_46428596470307
// MI455X (gfx1250) — matrix operand order repaired and checked by reading, not compiled, not run on hardware
//
#include <hip/hip_runtime.h>
#include <hip/hip_bf16.h>
#include <math.h>

// ---------------------------------------------------------------------------
// MAGNO layer for MI455X (gfx1250, wave32).
// GEMMs: v_wmma_f32_16x16x32_bf16, 32x32 wave tile (4 WMMA / K-step),
//        A tile staged via Tensor Data Mover (tensor_load_to_lds) when
//        available, B tile cooperatively transposed into LDS.
// Edge passes: global float atomics + int-punned float min/max atomics.
// ---------------------------------------------------------------------------

typedef __attribute__((ext_vector_type(16))) __bf16 v16bf;
typedef __attribute__((ext_vector_type(8)))  float  v8f;
typedef unsigned int u32x4 __attribute__((ext_vector_type(4)));
typedef int          i32x4 __attribute__((ext_vector_type(4)));
typedef int          i32x8 __attribute__((ext_vector_type(8)));

#if defined(__has_builtin)
#  if __has_builtin(__builtin_amdgcn_tensor_load_to_lds)
#    define HAVE_TDM 1
#  endif
#endif

#define D_DIM   256
#define H_NUM   8
#define HD_DIM  32
#define GEO_DIM 128
#define FFN_DIM 512
#define STAT_IN 12
#define LN_EPS  1e-5f

#define BT_M 128
#define BT_N 64
#define BT_K 32

__device__ __forceinline__ float gelu_erf(float x) {
    return 0.5f * x * (1.0f + erff(x * 0.70710678118654752f));
}

// float atomic min/max via monotonic int mapping (valid for mixed signs).
__device__ __forceinline__ void atomicMinF(float* p, float v) {
    if (v >= 0.0f) atomicMin((int*)p, __float_as_int(v));
    else           atomicMax((unsigned int*)p, __float_as_uint(v));
}
__device__ __forceinline__ void atomicMaxF(float* p, float v) {
    if (v >= 0.0f) atomicMax((int*)p, __float_as_int(v));
    else           atomicMin((unsigned int*)p, __float_as_uint(v));
}

// ---------------------------------------------------------------------------
// Utility kernels
// ---------------------------------------------------------------------------
__global__ void fill_kernel(float* __restrict__ p, float v, int n) {
    int i = blockIdx.x * blockDim.x + threadIdx.x;
    if (i < n) p[i] = v;
}

__global__ void f32_to_bf16_kernel(const float* __restrict__ in,
                                   __bf16* __restrict__ out, int n) {
    int i = blockIdx.x * blockDim.x + threadIdx.x;
    if (i < n) out[i] = (__bf16)in[i];
}

// stats layout per query: [0]=cnt [1..3]=sum [4..6]=sumsq [7..9]=min [10..12]=max
__global__ void init_stats_kernel(float* __restrict__ stats, int Q) {
    int q = blockIdx.x * blockDim.x + threadIdx.x;
    if (q >= Q) return;
    float* st = stats + q * 13;
    #pragma unroll
    for (int i = 0; i < 7; ++i) st[i] = 0.0f;
    #pragma unroll
    for (int i = 7; i < 10; ++i) st[i] = 1e30f;
    #pragma unroll
    for (int i = 10; i < 13; ++i) st[i] = -1e30f;
}

// ---------------------------------------------------------------------------
// Edge pass 1: segmented geo statistics
// ---------------------------------------------------------------------------
__global__ void edge_stats_kernel(const float* __restrict__ qpos,
                                  const float* __restrict__ spos,
                                  const int* __restrict__ q_idx,
                                  const int* __restrict__ s_idx,
                                  float* __restrict__ stats, int E) {
    int e = blockIdx.x * blockDim.x + threadIdx.x;
    if (e >= E) return;
    int q = q_idx[e], s = s_idx[e];
    float* st = stats + q * 13;
    atomicAdd(&st[0], 1.0f);
    #pragma unroll
    for (int c = 0; c < 3; ++c) {
        float r = spos[s * 3 + c] - qpos[q * 3 + c];
        atomicAdd(&st[1 + c], r);
        atomicAdd(&st[4 + c], r * r);
        atomicMinF(&st[7 + c], r);
        atomicMaxF(&st[10 + c], r);
    }
}

// ---------------------------------------------------------------------------
// Finalize stats + fused geo MLP: raw[12] -> gelu(.@Gw1+Gb1) -> gelu(.@Gw2+Gb2)
// ---------------------------------------------------------------------------
__global__ __launch_bounds__(GEO_DIM)
void geo_mlp_kernel(const float* __restrict__ stats,
                    const float* __restrict__ Gw1, const float* __restrict__ Gb1,
                    const float* __restrict__ Gw2, const float* __restrict__ Gb2,
                    __bf16* __restrict__ geo_bf) {
    __shared__ float raw[STAT_IN];
    __shared__ float h1[GEO_DIM];
    int q = blockIdx.x, j = threadIdx.x;
    if (j < STAT_IN) {
        const float* st = stats + q * 13;
        float cnt  = fmaxf(st[0], 1.0f);
        int kind = j / 3, c = j % 3;
        float mean = st[1 + c] / cnt;
        float val;
        if (kind == 0)      val = mean;
        else if (kind == 1) { float var = st[4 + c] / cnt - mean * mean;
                              val = sqrtf(fmaxf(var, 0.0f)); }
        else if (kind == 2) val = fminf(fmaxf(st[7 + c],  -100.0f), 100.0f);
        else                val = fminf(fmaxf(st[10 + c], -100.0f), 100.0f);
        raw[j] = val;
    }
    __syncthreads();
    float a = Gb1[j];
    #pragma unroll
    for (int i = 0; i < STAT_IN; ++i) a += raw[i] * Gw1[i * GEO_DIM + j];
    h1[j] = gelu_erf(a);
    __syncthreads();
    float b = Gb2[j];
    for (int k = 0; k < GEO_DIM; ++k) b += h1[k] * Gw2[k * GEO_DIM + j];
    geo_bf[q * GEO_DIM + j] = (__bf16)gelu_erf(b);
}

// ---------------------------------------------------------------------------
// LayerNorm over D=256, one block (256 threads) per row, bf16 output.
// ---------------------------------------------------------------------------
__global__ __launch_bounds__(D_DIM)
void ln_bf16_kernel(const float* __restrict__ X, const float* __restrict__ g,
                    const float* __restrict__ b, __bf16* __restrict__ Y) {
    __shared__ float red[D_DIM];
    int row = blockIdx.x, tid = threadIdx.x;
    float x = X[row * D_DIM + tid];
    red[tid] = x; __syncthreads();
    for (int s = D_DIM / 2; s > 0; s >>= 1) {
        if (tid < s) red[tid] += red[tid + s];
        __syncthreads();
    }
    float mean = red[0] / (float)D_DIM;
    __syncthreads();
    float d = x - mean;
    red[tid] = d * d; __syncthreads();
    for (int s = D_DIM / 2; s > 0; s >>= 1) {
        if (tid < s) red[tid] += red[tid + s];
        __syncthreads();
    }
    float var = red[0] / (float)D_DIM;
    float y = d * rsqrtf(var + LN_EPS) * g[tid] + b[tid];
    Y[row * D_DIM + tid] = (__bf16)y;
}

// ---------------------------------------------------------------------------
// bf16 WMMA GEMM: C[M,N] = A[M,K] x B[K,N]  (+bias)(+gelu)(+resid)
// Block tile 128x64, 8 waves (4M x 2N), wave tile 32x32 -> 4 WMMA / K-step.
// A tile (128x32) staged by the Tensor Data Mover; B tile (32x64)
// cooperatively transposed into LDS.  Layouts per CDNA5 ISA 7.12.2 (wave32).
// Requires M%128==0, N%64==0, K%32==0 (holds for every call here).
// ---------------------------------------------------------------------------
__global__ __launch_bounds__(256)
void gemm_bf16_wmma(const __bf16* __restrict__ A, const __bf16* __restrict__ B,
                    float* __restrict__ C, __bf16* __restrict__ Cbf,
                    const float* __restrict__ bias, const float* __restrict__ resid,
                    int M, int N, int K, int do_gelu) {
    __shared__ unsigned short sA[BT_M * BT_K];   // [m][k]  8KB
    __shared__ unsigned short sBt[BT_N * BT_K];  // [n][k]  4KB (transposed)
    const int m0 = blockIdx.y * BT_M;
    const int n0 = blockIdx.x * BT_N;
    const int tid  = threadIdx.x;
    const int wave = tid >> 5;
    const int lane = tid & 31;
    const int wy = wave & 3;          // M subtile 0..3 (32 rows each)
    const int wx = wave >> 2;         // N subtile 0..1 (32 cols each)
    const int lh = lane & 15;
    const int hi = (lane >> 4) & 1;

    v8f acc[2][2] = {};
    const unsigned int* Bd = (const unsigned int*)B;
#if !defined(HAVE_TDM)
    const unsigned int* Ad = (const unsigned int*)A;
#endif

    for (int k0 = 0; k0 < K; k0 += BT_K) {
        // ---- stage B tile transposed: 32x64 bf16 = 1024 dwords ----
        #pragma unroll
        for (int it = 0; it < 4; ++it) {
            int d = tid + it * 256;
            int r = d >> 5, c = d & 31;      // r = k-row 0..31, c = dword col
            unsigned int v = Bd[(((k0 + r) * N + n0) >> 1) + c];
            sBt[(2 * c + 0) * BT_K + r] = (unsigned short)(v & 0xffffu);
            sBt[(2 * c + 1) * BT_K + r] = (unsigned short)(v >> 16);
        }
#if defined(HAVE_TDM)
        // ---- stage A tile 128x32 via Tensor Data Mover (one wave issues) ----
        if (wave == 0) {
            unsigned long long ga =
                (unsigned long long)(uintptr_t)(A + (size_t)m0 * K + k0);
            unsigned int lds = (unsigned int)(uintptr_t)(void*)sA;
            u32x4 g0;
            g0[0] = 1u;                                   // count=1, user mode
            g0[1] = lds;                                  // lds_addr
            g0[2] = (unsigned int)(ga & 0xffffffffu);     // global_addr[31:0]
            g0[3] = (unsigned int)((ga >> 32) & 0x01ffffffu) | (2u << 30); // type=2
            i32x8 g1;
            g1[0] = 1 << 16;                              // data_size=1 (2B)
            g1[1] = (int)((K & 0xffff) << 16);            // tensor_dim0 lo
            g1[2] = (int)(((unsigned)K >> 16) & 0xffffu) |
                    (int)((M & 0xffff) << 16);            // dim0 hi | dim1 lo
            g1[3] = (int)(((unsigned)M >> 16) & 0xffffu) |
                    (BT_K << 16);                         // dim1 hi | tile_dim0=32
            g1[4] = BT_M;                                 // tile_dim1=128, tile_dim2=0
            g1[5] = K;                                    // dim0_stride lo
            g1[6] = 0;
            g1[7] = 0;
            i32x4 g2 = {0, 0, 0, 0};
            i32x4 g3 = {0, 0, 0, 0};
#if __clang_major__ >= 23
            i32x8 g4 = {0, 0, 0, 0, 0, 0, 0, 0};
            __builtin_amdgcn_tensor_load_to_lds(g0, g1, g2, g3, g4, 0);
#else
            __builtin_amdgcn_tensor_load_to_lds(g0, g1, g2, g3, 0);
#endif
            __builtin_amdgcn_s_wait_tensorcnt(0);
        }
#else
        // ---- cooperative A tile load: 128x32 bf16 = 2048 dwords ----
        #pragma unroll
        for (int it = 0; it < 8; ++it) {
            int d = tid + it * 256;
            int r = d >> 4, c = d & 15;
            ((unsigned int*)sA)[d] = Ad[(((m0 + r) * K + k0) >> 1) + c];
        }
        if (k0 + BT_K < K && tid < 128)
            __builtin_prefetch(A + (size_t)(m0 + tid) * K + k0 + BT_K, 0, 1);
#endif
        __syncthreads();

        // ---- fragments + 4 WMMA ----
        v16bf a0, a1, b0, b1;
        #pragma unroll
        for (int h = 0; h < 16; ++h) {
            // A frag: lanes 0-15 K {0-7,16-23}, lanes 16-31 K {8-15,24-31}
            int ka = h + (h & 8) + (hi ? 8 : 0);
            a0[h] = __builtin_bit_cast(__bf16, sA[(wy * 32 + lh) * BT_K + ka]);
            a1[h] = __builtin_bit_cast(__bf16, sA[(wy * 32 + 16 + lh) * BT_K + ka]);
            // B frag: lanes 0-15 K {0-7,16-23}, lanes 16-31 K {8-15,24-31}; N = lane&15
            int kb = h + (h & 8) + (hi ? 8 : 0);
            b0[h] = __builtin_bit_cast(__bf16, sBt[(wx * 32 + lh) * BT_K + kb]);
            b1[h] = __builtin_bit_cast(__bf16, sBt[(wx * 32 + 16 + lh) * BT_K + kb]);
        }
        acc[0][0] = __builtin_amdgcn_wmma_f32_16x16x32_bf16(
                        false, a0, false, b0, (short)0, acc[0][0], false, false);
        acc[0][1] = __builtin_amdgcn_wmma_f32_16x16x32_bf16(
                        false, a0, false, b1, (short)0, acc[0][1], false, false);
        acc[1][0] = __builtin_amdgcn_wmma_f32_16x16x32_bf16(
                        false, a1, false, b0, (short)0, acc[1][0], false, false);
        acc[1][1] = __builtin_amdgcn_wmma_f32_16x16x32_bf16(
                        false, a1, false, b1, (short)0, acc[1][1], false, false);
        __syncthreads();
    }

    // ---- epilogue:  C layout: VGPR r -> M = r + (hi?8:0), N = lane&15 ----
    #pragma unroll
    for (int i = 0; i < 2; ++i) {
        const int mbase = m0 + wy * 32 + i * 16 + (hi ? 8 : 0);
        #pragma unroll
        for (int j = 0; j < 2; ++j) {
            const int colg = n0 + wx * 32 + j * 16 + lh;
            const float bval = bias ? bias[colg] : 0.0f;
            #pragma unroll
            for (int r = 0; r < 8; ++r) {
                int mg = mbase + r;
                float v = acc[i][j][r] + bval;
                if (do_gelu) v = gelu_erf(v);
                if (resid)   v += resid[(size_t)mg * N + colg];
                if (C)       C[(size_t)mg * N + colg] = v;
                if (Cbf)     Cbf[(size_t)mg * N + colg] = (__bf16)v;
            }
        }
    }
}

// ---------------------------------------------------------------------------
// Attention edge passes (one wave per edge; lane covers 8 contiguous channels)
// ---------------------------------------------------------------------------
__global__ __launch_bounds__(256)
void edge_scores_kernel(const float* __restrict__ Qf, const float* __restrict__ Kf,
                        const int* __restrict__ q_idx, const int* __restrict__ s_idx,
                        const float* __restrict__ log_tau,
                        float* __restrict__ scores, float* __restrict__ smax, int E) {
    int e = blockIdx.x * 8 + (threadIdx.x >> 5);
    if (e >= E) return;
    int lane = threadIdx.x & 31;
    int q = q_idx[e], s = s_idx[e];
    const float* qp = Qf + (size_t)q * D_DIM + lane * 8;
    const float* kp = Kf + (size_t)s * D_DIM + lane * 8;
    float p = 0.0f;
    #pragma unroll
    for (int i = 0; i < 8; ++i) p += qp[i] * kp[i];
    p += __shfl_xor(p, 1, 32);
    p += __shfl_xor(p, 2, 32);      // 4 lanes = one head (HD=32)
    if ((lane & 3) == 0) {
        float tau = expf(log_tau[0]);
        float sc = p * (0.17677669529663687f / tau);   // 1/sqrt(32)/tau
        int h = lane >> 2;
        scores[(size_t)e * H_NUM + h] = sc;
        if (sc > 0.0f)   // smax init = 0 (torch max starts at 0); positives only
            atomicMax((int*)&smax[q * H_NUM + h], __float_as_int(sc));
    }
}

__global__ void edge_exp_kernel(const int* __restrict__ q_idx,
                                const float* __restrict__ smax,
                                float* __restrict__ scores,
                                float* __restrict__ ssum, int E) {
    int t = blockIdx.x * blockDim.x + threadIdx.x;
    if (t >= E * H_NUM) return;
    int e = t >> 3, h = t & 7;
    int q = q_idx[e];
    float ex = expf(scores[t] - smax[q * H_NUM + h]);
    scores[t] = ex;
    atomicAdd(&ssum[q * H_NUM + h], ex);
}

__global__ __launch_bounds__(256)
void edge_weighted_kernel(const float* __restrict__ Vf, const float* __restrict__ Gf,
                          const int* __restrict__ q_idx, const int* __restrict__ s_idx,
                          const float* __restrict__ scores, const float* __restrict__ ssum,
                          float* __restrict__ attn, int E) {
    int e = blockIdx.x * 8 + (threadIdx.x >> 5);
    if (e >= E) return;
    int lane = threadIdx.x & 31;
    int q = q_idx[e], s = s_idx[e];
    int h = lane >> 2;
    float alpha = scores[(size_t)e * H_NUM + h] /
                  fmaxf(ssum[q * H_NUM + h], 1e-8f);
    const float* vp = Vf + (size_t)s * D_DIM + lane * 8;
    const float* gp = Gf + (size_t)q * D_DIM + lane * 8;
    float* op = attn + (size_t)q * D_DIM + lane * 8;
    #pragma unroll
    for (int i = 0; i < 8; ++i) atomicAdd(&op[i], alpha * (vp[i] + gp[i]));
}

// ---------------------------------------------------------------------------
// Host-side orchestration
// ---------------------------------------------------------------------------
extern "C" void kernel_launch(void* const* d_in, const int* in_sizes, int n_in,
                              void* d_out, int out_size, void* d_ws, size_t ws_size,
                              hipStream_t stream) {
    const float* query_tokens  = (const float*)d_in[0];
    const float* query_pos     = (const float*)d_in[1];
    const float* support_feats = (const float*)d_in[2];
    const float* support_pos   = (const float*)d_in[3];
    const float* Wq  = (const float*)d_in[4];
    const float* Wk  = (const float*)d_in[5];
    const float* Wv  = (const float*)d_in[6];
    const float* Wg  = (const float*)d_in[7];
    const float* Wo  = (const float*)d_in[8];
    const float* bo  = (const float*)d_in[9];
    const float* log_tau = (const float*)d_in[10];
    const float* ln1_g = (const float*)d_in[11];
    const float* ln1_b = (const float*)d_in[12];
    const float* ln2_g = (const float*)d_in[13];
    const float* ln2_b = (const float*)d_in[14];
    const float* Wf1 = (const float*)d_in[15];
    const float* bf1 = (const float*)d_in[16];
    const float* Wf2 = (const float*)d_in[17];
    const float* bf2 = (const float*)d_in[18];
    const float* Gw1 = (const float*)d_in[19];
    const float* Gb1 = (const float*)d_in[20];
    const float* Gw2 = (const float*)d_in[21];
    const float* Gb2 = (const float*)d_in[22];
    const int* q_idx = (const int*)d_in[23];
    const int* s_idx = (const int*)d_in[24];

    const int Q = in_sizes[0] / D_DIM;
    const int S = in_sizes[2] / D_DIM;
    const int E = in_sizes[23];
    float* out = (float*)d_out;

    // workspace carve-out (256B aligned regions)
    char* base = (char*)d_ws;
    size_t off = 0;
    auto alloc = [&](size_t bytes) -> char* {
        char* p = base + off;
        off += (bytes + 255) & ~(size_t)255;
        return p;
    };
    float*  stats   = (float*) alloc((size_t)Q * 13 * 4);
    __bf16* geo_bf  = (__bf16*)alloc((size_t)Q * GEO_DIM * 2);
    __bf16* qt_bf   = (__bf16*)alloc((size_t)Q * D_DIM * 2);
    __bf16* sf_bf   = (__bf16*)alloc((size_t)S * D_DIM * 2);
    __bf16* Wq_bf   = (__bf16*)alloc((size_t)D_DIM * D_DIM * 2);
    __bf16* Wk_bf   = (__bf16*)alloc((size_t)D_DIM * D_DIM * 2);
    __bf16* Wv_bf   = (__bf16*)alloc((size_t)D_DIM * D_DIM * 2);
    __bf16* Wg_bf   = (__bf16*)alloc((size_t)GEO_DIM * D_DIM * 2);
    __bf16* Wo_bf   = (__bf16*)alloc((size_t)D_DIM * D_DIM * 2);
    __bf16* Wf1_bf  = (__bf16*)alloc((size_t)D_DIM * FFN_DIM * 2);
    __bf16* Wf2_bf  = (__bf16*)alloc((size_t)FFN_DIM * D_DIM * 2);
    float*  Qf      = (float*) alloc((size_t)Q * D_DIM * 4);
    float*  Kf      = (float*) alloc((size_t)S * D_DIM * 4);
    float*  Vf      = (float*) alloc((size_t)S * D_DIM * 4);
    float*  Gf      = (float*) alloc((size_t)Q * D_DIM * 4);
    float*  scores  = (float*) alloc((size_t)E * H_NUM * 4);
    float*  smax    = (float*) alloc((size_t)Q * H_NUM * 4);
    float*  ssum    = (float*) alloc((size_t)Q * H_NUM * 4);
    float*  attn    = (float*) alloc((size_t)Q * D_DIM * 4);
    __bf16* attn_bf = (__bf16*)alloc((size_t)Q * D_DIM * 2);
    float*  xres    = (float*) alloc((size_t)Q * D_DIM * 4);
    __bf16* h2_bf   = (__bf16*)alloc((size_t)Q * D_DIM * 2);
    __bf16* f1_bf   = (__bf16*)alloc((size_t)Q * FFN_DIM * 2);

    // ---- init accumulators ----
    init_stats_kernel<<<(Q + 255) / 256, 256, 0, stream>>>(stats, Q);
    fill_kernel<<<(Q * H_NUM + 255) / 256, 256, 0, stream>>>(smax, 0.0f, Q * H_NUM);
    fill_kernel<<<(Q * H_NUM + 255) / 256, 256, 0, stream>>>(ssum, 0.0f, Q * H_NUM);
    fill_kernel<<<(Q * D_DIM + 255) / 256, 256, 0, stream>>>(attn, 0.0f, Q * D_DIM);

    // ---- geo stats + MLP ----
    edge_stats_kernel<<<(E + 255) / 256, 256, 0, stream>>>(
        query_pos, support_pos, q_idx, s_idx, stats, E);
    geo_mlp_kernel<<<Q, GEO_DIM, 0, stream>>>(stats, Gw1, Gb1, Gw2, Gb2, geo_bf);

    // ---- precision conversions ----
    ln_bf16_kernel<<<Q, D_DIM, 0, stream>>>(query_tokens, ln1_g, ln1_b, qt_bf);
    f32_to_bf16_kernel<<<(S * D_DIM + 255) / 256, 256, 0, stream>>>(support_feats, sf_bf, S * D_DIM);
    f32_to_bf16_kernel<<<(D_DIM * D_DIM + 255) / 256, 256, 0, stream>>>(Wq, Wq_bf, D_DIM * D_DIM);
    f32_to_bf16_kernel<<<(D_DIM * D_DIM + 255) / 256, 256, 0, stream>>>(Wk, Wk_bf, D_DIM * D_DIM);
    f32_to_bf16_kernel<<<(D_DIM * D_DIM + 255) / 256, 256, 0, stream>>>(Wv, Wv_bf, D_DIM * D_DIM);
    f32_to_bf16_kernel<<<(GEO_DIM * D_DIM + 255) / 256, 256, 0, stream>>>(Wg, Wg_bf, GEO_DIM * D_DIM);
    f32_to_bf16_kernel<<<(D_DIM * D_DIM + 255) / 256, 256, 0, stream>>>(Wo, Wo_bf, D_DIM * D_DIM);
    f32_to_bf16_kernel<<<(D_DIM * FFN_DIM + 255) / 256, 256, 0, stream>>>(Wf1, Wf1_bf, D_DIM * FFN_DIM);
    f32_to_bf16_kernel<<<(FFN_DIM * D_DIM + 255) / 256, 256, 0, stream>>>(Wf2, Wf2_bf, FFN_DIM * D_DIM);

    // ---- projection GEMMs (WMMA bf16) ----
    gemm_bf16_wmma<<<dim3(D_DIM / BT_N, Q / BT_M), 256, 0, stream>>>(
        qt_bf, Wq_bf, Qf, nullptr, nullptr, nullptr, Q, D_DIM, D_DIM, 0);
    gemm_bf16_wmma<<<dim3(D_DIM / BT_N, S / BT_M), 256, 0, stream>>>(
        sf_bf, Wk_bf, Kf, nullptr, nullptr, nullptr, S, D_DIM, D_DIM, 0);
    gemm_bf16_wmma<<<dim3(D_DIM / BT_N, S / BT_M), 256, 0, stream>>>(
        sf_bf, Wv_bf, Vf, nullptr, nullptr, nullptr, S, D_DIM, D_DIM, 0);
    gemm_bf16_wmma<<<dim3(D_DIM / BT_N, Q / BT_M), 256, 0, stream>>>(
        geo_bf, Wg_bf, Gf, nullptr, nullptr, nullptr, Q, D_DIM, GEO_DIM, 0);

    // ---- edge attention ----
    edge_scores_kernel<<<E / 8, 256, 0, stream>>>(
        Qf, Kf, q_idx, s_idx, log_tau, scores, smax, E);
    edge_exp_kernel<<<(E * H_NUM + 255) / 256, 256, 0, stream>>>(
        q_idx, smax, scores, ssum, E);
    edge_weighted_kernel<<<E / 8, 256, 0, stream>>>(
        Vf, Gf, q_idx, s_idx, scores, ssum, attn, E);

    // ---- output projection + residual: x = query_tokens + attn@Wo + bo ----
    f32_to_bf16_kernel<<<(Q * D_DIM + 255) / 256, 256, 0, stream>>>(attn, attn_bf, Q * D_DIM);
    gemm_bf16_wmma<<<dim3(D_DIM / BT_N, Q / BT_M), 256, 0, stream>>>(
        attn_bf, Wo_bf, xres, nullptr, bo, query_tokens, Q, D_DIM, D_DIM, 0);

    // ---- FFN: out = x + gelu(LN2(x)@Wf1+bf1)@Wf2 + bf2 ----
    ln_bf16_kernel<<<Q, D_DIM, 0, stream>>>(xres, ln2_g, ln2_b, h2_bf);
    gemm_bf16_wmma<<<dim3(FFN_DIM / BT_N, Q / BT_M), 256, 0, stream>>>(
        h2_bf, Wf1_bf, nullptr, f1_bf, bf1, nullptr, Q, FFN_DIM, D_DIM, 1);
    gemm_bf16_wmma<<<dim3(D_DIM / BT_N, Q / BT_M), 256, 0, stream>>>(
        f1_bf, Wf2_bf, out, nullptr, bf2, xres, Q, D_DIM, FFN_DIM, 0);
}
